// CrossAttention_13872744366837
// MI455X (gfx1250) — matrix operand order repaired and checked by reading, not compiled, not run on hardware
//
#include <hip/hip_runtime.h>
#include <hip/hip_bf16.h>

// ---------------------------------------------------------------------------
// CrossAttention for MI455X (gfx1250): bf16 WMMA pipeline, f32 accumulate.
// B=2, L=2048, DIM=2048, H=16, Dh=128.  SCALE = 1/sqrt(128).
// Weights pre-transposed once so all hot-loop staging is k-contiguous copies;
// V written head-transposed so attention stages only P (not K/V) in LDS.
// GEMM tiles staged with GLOBAL_LOAD_ASYNC_TO_LDS_B128, double-buffered so
// the async DMA of tile k+1 overlaps the 16 WMMAs of tile k.
// ---------------------------------------------------------------------------

typedef __bf16 bf16;
typedef __attribute__((ext_vector_type(16))) __bf16 v16bf;
typedef __attribute__((ext_vector_type(8)))  __bf16 v8bf;
typedef __attribute__((ext_vector_type(4)))  __bf16 v4bf;
typedef __attribute__((ext_vector_type(8)))  float  v8f;
typedef __attribute__((ext_vector_type(4)))  int    v4i;

#define ATT_SCALE 0.08838834764831845f  // 128^-0.5
#define L_SEQ 2048
#define DHEAD 128
#define NHEAD 16

union AFrag { v16bf v; v8bf h[2]; };

#if defined(__has_builtin)
#if __has_builtin(__builtin_amdgcn_global_load_async_to_lds_b128)
#define HAVE_ASYNC_LDS 1
#endif
#endif
#ifndef HAVE_ASYNC_LDS
#define HAVE_ASYNC_LDS 0
#endif
// Builtin signature (probe-derived): (v4i as1* src, v4i as3* dst, imm, imm cpol)
#define GAS1(p) ((__attribute__((address_space(1))) v4i*)(p))
#define LAS3(p) ((__attribute__((address_space(3))) v4i*)(p))

__device__ __forceinline__ void wait_async_all() {
#if defined(__has_builtin) && __has_builtin(__builtin_amdgcn_s_wait_asynccnt)
    __builtin_amdgcn_s_wait_asynccnt(0);
#else
    asm volatile("s_wait_asynccnt 0" ::: "memory");
#endif
}

// ---------------------------------------------------------------------------
// fp32 -> bf16 conversion (4 elements / thread)
// ---------------------------------------------------------------------------
__global__ void cvt_f32_bf16(const float* __restrict__ s, bf16* __restrict__ d, int n4) {
    int i = blockIdx.x * blockDim.x + threadIdx.x;
    if (i < n4) {
        float4 v = ((const float4*)s)[i];
        v4bf o;
        o[0] = (bf16)v.x; o[1] = (bf16)v.y; o[2] = (bf16)v.z; o[3] = (bf16)v.w;
        ((v4bf*)d)[i] = o;
    }
}

// ---------------------------------------------------------------------------
// fp32 [R][C] -> bf16 [C][R] transpose (one-time weight prep), LDS tiled.
// ---------------------------------------------------------------------------
__global__ __launch_bounds__(256) void cvt_transpose(
    const float* __restrict__ src, bf16* __restrict__ dst, int R, int C) {
    __shared__ bf16 tile[32][33];
    const int bx = blockIdx.x * 32, by = blockIdx.y * 32;
    const int tx = threadIdx.x, ty = threadIdx.y;   // (32, 8)
#pragma unroll
    for (int j = 0; j < 32; j += 8)
        tile[ty + j][tx] = (bf16)src[(size_t)(by + ty + j) * C + bx + tx];
    __syncthreads();
#pragma unroll
    for (int j = 0; j < 32; j += 8)
        dst[(size_t)(bx + ty + j) * R + by + tx] = tile[tx][ty + j];
}

// ---------------------------------------------------------------------------
// Tiled bf16 GEMM with bias:  C[M x N] = A[M x K] @ BT[N x K]^T + bias[N]
// (B supplied pre-transposed, row stride K -> both tiles stage k-contiguous.)
// 128 threads (4 waves, 2x2), block tile 128x128, wave tile 64x64, K-step 32.
// Double-buffered async global->LDS staging: DMA of tile k+1 overlaps the
// 16 WMMAs of tile k.
// mode 0: bf16 head-scatter -> dst0 = Q[(b*16+h)*2048 + l][128]
// mode 1: N=4096; cols <2048 -> dst0 (K, head layout),
//                 cols >=2048 -> dst1 (V, head-TRANSPOSED layout [bh][d][key])
// mode 2: f32 row-major -> dstf (final projection output)
// ---------------------------------------------------------------------------
__global__ __launch_bounds__(128) void gemm_bf16_wmma(
    const bf16* __restrict__ A, const bf16* __restrict__ BT,
    const float* __restrict__ bias,
    bf16* __restrict__ dst0, bf16* __restrict__ dst1, float* __restrict__ dstf,
    int M, int N, int K, int mode)
{
    __shared__ __align__(32) bf16 sA[2][128][32];    // A tiles  [buf][m][k]
    __shared__ __align__(32) bf16 sBT[2][128][32];   // B tiles  [buf][n][k]

    const int t    = threadIdx.x;
    const int lane = t & 31;
    const int wave = t >> 5;
    const int lh   = lane >> 4;
    const int ll   = lane & 15;
    const int m0   = blockIdx.y * 128;
    const int n0   = blockIdx.x * 128;
    const int wm   = (wave >> 1) * 64;
    const int wn   = (wave & 1) * 64;

    const bf16* gaRow = A  + (size_t)(m0 + t) * K;   // this thread's 64B rows
    const bf16* gbRow = BT + (size_t)(n0 + t) * K;

    // stage one 32-wide K-slice of both tiles into LDS buffer `buf`
    auto stage = [&](int buf, int k0) {
        const bf16* ga = gaRow + k0;
        const bf16* gb = gbRow + k0;
#if HAVE_ASYNC_LDS
        __builtin_amdgcn_global_load_async_to_lds_b128(GAS1(ga),      LAS3(&sA[buf][t][0]),   0, 0);
        __builtin_amdgcn_global_load_async_to_lds_b128(GAS1(ga + 8),  LAS3(&sA[buf][t][8]),   0, 0);
        __builtin_amdgcn_global_load_async_to_lds_b128(GAS1(ga + 16), LAS3(&sA[buf][t][16]),  0, 0);
        __builtin_amdgcn_global_load_async_to_lds_b128(GAS1(ga + 24), LAS3(&sA[buf][t][24]),  0, 0);
        __builtin_amdgcn_global_load_async_to_lds_b128(GAS1(gb),      LAS3(&sBT[buf][t][0]),  0, 0);
        __builtin_amdgcn_global_load_async_to_lds_b128(GAS1(gb + 8),  LAS3(&sBT[buf][t][8]),  0, 0);
        __builtin_amdgcn_global_load_async_to_lds_b128(GAS1(gb + 16), LAS3(&sBT[buf][t][16]), 0, 0);
        __builtin_amdgcn_global_load_async_to_lds_b128(GAS1(gb + 24), LAS3(&sBT[buf][t][24]), 0, 0);
#else
        *(v16bf*)&sA[buf][t][0]   = *(const v16bf*)ga;
        *(v16bf*)&sA[buf][t][16]  = *(const v16bf*)(ga + 16);
        *(v16bf*)&sBT[buf][t][0]  = *(const v16bf*)gb;
        *(v16bf*)&sBT[buf][t][16] = *(const v16bf*)(gb + 16);
#endif
    };

    v8f acc[4][4] = {};

    // prologue: fill buffer 0
    stage(0, 0);
    wait_async_all();
    __syncthreads();

    for (int k0 = 0; k0 < K; k0 += 32) {
        const int  cur  = (k0 >> 5) & 1;
        const bool more = (k0 + 32) < K;

        // kick off DMA for the next tile into the other buffer (overlaps WMMA)
        if (more) stage(cur ^ 1, k0 + 32);

        // ---- fragment gathers (16-bit WMMA register layouts) ----
        AFrag af[4]; v16bf bfg[4];
#pragma unroll
        for (int i = 0; i < 4; ++i) {
            int m = wm + i * 16 + ll;
            af[i].v = *(const v16bf*)&sA[cur][m][lh * 16];
        }
#pragma unroll
        for (int j = 0; j < 4; ++j) {
            int n = wn + j * 16 + ll;
            bfg[j] = *(const v16bf*)&sBT[cur][n][lh * 16];
        }
#pragma unroll
        for (int i = 0; i < 4; ++i)
#pragma unroll
            for (int j = 0; j < 4; ++j)
                acc[i][j] = __builtin_amdgcn_wmma_f32_16x16x32_bf16(
                    false, af[i].v, false, bfg[j], (short)0, acc[i][j], false, false);

        if (more) wait_async_all();   // next buffer landed
        __syncthreads();              // everyone done reading `cur`
    }

    // ---- epilogue: bias + scatter ----
#pragma unroll
    for (int i = 0; i < 4; ++i)
#pragma unroll
        for (int j = 0; j < 4; ++j)
#pragma unroll
            for (int r = 0; r < 8; ++r) {
                int m = m0 + wm + i * 16 + r + lh * 8;   // C layout: M = r + half*8
                int n = n0 + wn + j * 16 + ll;           // C layout: N = lane&15
                float v = acc[i][j][r] + bias[n];
                if (mode == 2) {
                    dstf[(size_t)m * N + n] = v;
                } else {
                    int b = m >> 11, l = m & 2047;
                    if (mode == 1 && n >= 2048) {
                        int nn = n - 2048, h = nn >> 7, d = nn & 127;
                        // V: head-transposed [bh][d][key]
                        dst1[(((size_t)(b * NHEAD + h)) * DHEAD + d) * L_SEQ + l] = (bf16)v;
                    } else {
                        int h = n >> 7, d = n & 127;
                        dst0[((((size_t)b * NHEAD + h) * L_SEQ) + l) * DHEAD + d] = (bf16)v;
                    }
                }
            }
}

// ---------------------------------------------------------------------------
// Flash attention: grid (L/64, B*H), 128 threads (4 independent waves).
// Wave: 16 query rows x 128 head-dim; keys in 32-wide blocks.
// K and V^T fragments load straight from global (32B contiguous per lane).
// No workgroup barriers.
// ---------------------------------------------------------------------------
__global__ __launch_bounds__(128) void attn_wmma(
    const bf16* __restrict__ Qh, const bf16* __restrict__ Kh,
    const bf16* __restrict__ Vth, bf16* __restrict__ O)
{
    __shared__ __align__(32) bf16 sP[4][16][32];  // per-wave P staging (1KB each)

    const int t    = threadIdx.x;
    const int lane = t & 31;
    const int wave = t >> 5;
    const int lh   = lane >> 4;
    const int ll   = lane & 15;
    const int bh   = blockIdx.y;               // b*16+h
    const int q0   = blockIdx.x * 64 + wave * 16;
    const int b    = bh >> 4, h = bh & 15;

    const bf16* Q  = Qh  + (size_t)bh * L_SEQ * DHEAD;
    const bf16* Kp = Kh  + (size_t)bh * L_SEQ * DHEAD;
    const bf16* Vt = Vth + (size_t)bh * DHEAD * L_SEQ;   // [d][key]

    // Q tile 16x128 as 4 A-fragments, resident in VGPRs
    AFrag qf[4];
#pragma unroll
    for (int c = 0; c < 4; ++c) {
        const bf16* qrow = Q + (size_t)(q0 + ll) * DHEAD + c * 32 + lh * 16;
        qf[c].v = *(const v16bf*)qrow;
    }

    v8f acc[8] = {};
    float rmax[8], rsum[8];
#pragma unroll
    for (int r = 0; r < 8; ++r) { rmax[r] = -1e30f; rsum[r] = 0.f; }

    for (int kb = 0; kb < L_SEQ; kb += 32) {
        if (kb + 32 < L_SEQ) {                      // global_prefetch_b8
            __builtin_prefetch(Kp + (size_t)(kb + 32) * DHEAD, 0, 0);
            __builtin_prefetch(Vt + (size_t)(ll * 16) * L_SEQ + kb + 32, 0, 0);
        }

        // ---- S(16x32) = Q K^T : 2 n-tiles x 4 k-chunks ----
        v8f s[2] = {};
#pragma unroll
        for (int tt = 0; tt < 2; ++tt) {
            int key = kb + tt * 16 + ll;            // B-frag col = lane&15
#pragma unroll
            for (int c = 0; c < 4; ++c) {
                v16bf kf = *(const v16bf*)(Kp + (size_t)key * DHEAD + c * 32 + lh * 16);
                s[tt] = __builtin_amdgcn_wmma_f32_16x16x32_bf16(
                    false, qf[c].v, false, kf, (short)0, s[tt], false, false);
            }
        }
#pragma unroll
        for (int tt = 0; tt < 2; ++tt)
#pragma unroll
            for (int r = 0; r < 8; ++r) s[tt][r] *= ATT_SCALE;

        // ---- online softmax across 32 keys (16-lane butterflies) ----
#pragma unroll
        for (int r = 0; r < 8; ++r) {
            float mx = fmaxf(s[0][r], s[1][r]);
#pragma unroll
            for (int off = 8; off >= 1; off >>= 1)
                mx = fmaxf(mx, __shfl_xor(mx, off, 32));
            float nm   = fmaxf(rmax[r], mx);
            float corr = __expf(rmax[r] - nm);
            float p0 = __expf(s[0][r] - nm);
            float p1 = __expf(s[1][r] - nm);
            s[0][r] = p0; s[1][r] = p1;
            float ps = p0 + p1;
#pragma unroll
            for (int off = 8; off >= 1; off >>= 1)
                ps += __shfl_xor(ps, off, 32);
            rsum[r] = rsum[r] * corr + ps;
            rmax[r] = nm;
#pragma unroll
            for (int dt = 0; dt < 8; ++dt) acc[dt][r] *= corr;
        }

        // ---- P: C-layout -> A-layout via wave-private LDS round trip ----
        bf16* Pw = &sP[wave][0][0];
#pragma unroll
        for (int r = 0; r < 8; ++r) {
            int m = r + lh * 8;
            Pw[m * 32 + ll]      = (bf16)s[0][r];
            Pw[m * 32 + 16 + ll] = (bf16)s[1][r];
        }
        asm volatile("s_wait_dscnt 0" ::: "memory");  // wave-internal LDS fence
        AFrag pf;
        {
            const bf16* pr = Pw + (size_t)ll * 32 + lh * 16;
            pf.v = *(const v16bf*)pr;
        }

        // ---- O(16x128) += P(16x32) V(32x128): 8 WMMAs, V^T direct global ----
#pragma unroll
        for (int dt = 0; dt < 8; ++dt) {
            v16bf vf = *(const v16bf*)(Vt + (size_t)(dt * 16 + ll) * L_SEQ + kb + lh * 16);
            acc[dt] = __builtin_amdgcn_wmma_f32_16x16x32_bf16(
                false, pf.v, false, vf, (short)0, acc[dt], false, false);
        }
    }

    // ---- normalize + store O as [B, L, DIM] bf16 ----
#pragma unroll
    for (int r = 0; r < 8; ++r) {
        float inv = 1.0f / rsum[r];
        int m = q0 + r + lh * 8;
#pragma unroll
        for (int dt = 0; dt < 8; ++dt) {
            int d = dt * 16 + ll;
            O[((size_t)b * L_SEQ + m) * 2048 + h * DHEAD + d] = (bf16)(acc[dt][r] * inv);
        }
    }
}

// ---------------------------------------------------------------------------
// Host launch
// ---------------------------------------------------------------------------
extern "C" void kernel_launch(void* const* d_in, const int* in_sizes, int n_in,
                              void* d_out, int out_size, void* d_ws, size_t ws_size,
                              hipStream_t stream) {
    const float* x     = (const float*)d_in[0];   // [2,2048,2048]
    const float* ctx   = (const float*)d_in[1];   // [2,2048,2048]
    const float* Wqkv  = (const float*)d_in[2];   // [2048,6144]
    const float* bqkv  = (const float*)d_in[3];   // [6144]
    const float* Wproj = (const float*)d_in[4];   // [2048,2048]
    const float* bproj = (const float*)d_in[5];   // [2048]
    float* out = (float*)d_out;

    const size_t N_X   = 2u * 2048u * 2048u;   // 8,388,608
    const size_t N_QKV = 2048u * 6144u;        // 12,582,912
    const size_t N_PRJ = 2048u * 2048u;        // 4,194,304

    bf16* xb     = (bf16*)d_ws;
    bf16* cb     = xb + N_X;
    bf16* wqkvT  = cb + N_X;        // [6144][2048] (transposed)
    bf16* wprojT = wqkvT + N_QKV;   // [2048][2048] (transposed)
    bf16* Qh     = wprojT + N_PRJ;  // [32][2048][128]
    bf16* Kh     = Qh + N_X;        // [32][2048][128]
    bf16* Vt     = Kh + N_X;        // [32][128][2048] (head-transposed)
    bf16* Ob     = Vt + N_X;        // [4096][2048]    (~128 MB total)

    // 1) fp32 -> bf16 (+ weight transposes, one time)
    auto cvt = [&](const float* s, bf16* d, size_t n) {
        int n4 = (int)(n / 4);
        cvt_f32_bf16<<<(n4 + 255) / 256, 256, 0, stream>>>(s, d, n4);
    };
    cvt(x, xb, N_X);
    cvt(ctx, cb, N_X);
    cvt_transpose<<<dim3(6144 / 32, 2048 / 32), dim3(32, 8), 0, stream>>>(
        Wqkv, wqkvT, 2048, 6144);
    cvt_transpose<<<dim3(2048 / 32, 2048 / 32), dim3(32, 8), 0, stream>>>(
        Wproj, wprojT, 2048, 2048);

    // 2) Q = x @ Wqkv[:, :2048] + b   (BT rows 0..2047)
    gemm_bf16_wmma<<<dim3(2048 / 128, 4096 / 128), 128, 0, stream>>>(
        xb, wqkvT, bqkv, Qh, nullptr, nullptr, 4096, 2048, 2048, /*mode=*/0);

    //    KV = ctx @ Wqkv[:, 2048:] + b  (BT rows 2048..6143) -> K head, V head^T
    gemm_bf16_wmma<<<dim3(4096 / 128, 4096 / 128), 128, 0, stream>>>(
        cb, wqkvT + (size_t)2048 * 2048, bqkv + 2048, Kh, Vt, nullptr,
        4096, 4096, 2048, /*mode=*/1);

    // 3) attention -> Ob [B, L, DIM] bf16
    attn_wmma<<<dim3(2048 / 64, 32), 128, 0, stream>>>(Qh, Kh, Vt, Ob);

    // 4) out = Ob @ Wproj + b_proj (fp32)
    gemm_bf16_wmma<<<dim3(2048 / 128, 4096 / 128), 128, 0, stream>>>(
        Ob, wprojT, bproj, nullptr, nullptr, out, 4096, 2048, 2048, /*mode=*/2);
}
